// Net_79602923864556
// MI455X (gfx1250) — matrix operand order repaired and checked by reading, not compiled, not run on hardware
//
#include <hip/hip_runtime.h>

// CDNA5 / gfx1250, wave32. All matrix math via V_WMMA_F32_16X16X32_F16.
// Single fused kernel: one workgroup per batch element, all intermediates in
// LDS (~55 KB of the 320 KB/WGP), zero workspace, one launch total.
typedef __attribute__((ext_vector_type(16))) _Float16 v16h;
typedef __attribute__((ext_vector_type(8)))  float    v8f;

constexpr int B_ = 32, N_ = 116, E_ = 1346;
constexpr int IN_V = 116, IN_E = 5, H_ = 64, OUT_ = 4;
constexpr float NEG_SLOPE = 0.01f;
constexpr int NCH  = (E_ + 31) / 32;  // 43 K-chunks over the edge dimension
constexpr int EPAD = NCH * 32;        // 1376

// ---- WMMA fragment loaders (ISA 7.12.2 layouts, wave32) ----------------
// A (16-bit, 16xK tile at k0): lane l -> row M=l&15; lanes>=16 shift K by +8;
// halves: VGPR v holds K pairs {2v,2v+1} (v<4) / {16+2(v-4),...} (v>=4).
__device__ __forceinline__ v16h load_a_frag(const _Float16* lds, int row0,
                                            int k0, int ld) {
  int lane = threadIdx.x & 31;
  const _Float16* p =
      lds + (size_t)(row0 + (lane & 15)) * ld + k0 + ((lane >> 4) << 3);
  v16h a;
#pragma unroll
  for (int i = 0; i < 16; ++i) {
    int v = i >> 1;
    int k = ((v & 3) << 1) + (i & 1) + ((v >> 2) << 4);
    a[i] = p[k];
  }
  return a;
}

// B (16-bit, Kx16): lane l -> N=col0+(l&15); lanes>=16 shift K by +8; K order as A.
// Source stored row-major [K][N] with stride ld.
__device__ __forceinline__ v16h load_b_frag_KN(const _Float16* lds, int col0,
                                               int k0, int ld) {
  int lane = threadIdx.x & 31;
  const _Float16* p =
      lds + (size_t)(k0 + ((lane >> 4) << 3)) * ld + col0 + (lane & 15);
  v16h b;
#pragma unroll
  for (int i = 0; i < 16; ++i) {
    int v = i >> 1;
    int k = ((v & 3) << 1) + (i & 1) + ((v >> 2) << 4);
    b[i] = p[(size_t)k * ld];
  }
  return b;
}

// Same B fragment but source stored transposed [N][K] (for T^T).
__device__ __forceinline__ v16h load_b_frag_NK(const _Float16* lds, int col0,
                                               int ld) {
  int lane = threadIdx.x & 31;
  v16h b;
#pragma unroll
  for (int i = 0; i < 16; ++i) {
    int v = i >> 1;
    int k = ((v & 3) << 1) + (i & 1) + ((v >> 2) << 4);
    b[i] = lds[(size_t)(col0 + (lane & 15)) * ld + ((lane >> 4) << 3) + k];
  }
  return b;
}

// ---- fused node-layer network (edge layer is dead w.r.t. the output) ----
__global__ void __launch_bounds__(256)
k_fused(const float* __restrict__ Hvin, const float* __restrict__ Hein,
        const float* __restrict__ Av, const float* __restrict__ T,
        const float* __restrict__ Wv, const float* __restrict__ pv,
        const float* __restrict__ bv, const float* __restrict__ Wfc,
        const float* __restrict__ bfc, float* __restrict__ out) {
  // 48 KB phase-overlaid arena + small persistent arrays.
  __shared__ unsigned char smem[49152];
  __shared__ float de_s[EPAD];   // diag(de) for this batch, zero-padded
  __shared__ float pool_s[H_];   // pooled accumulator

  const int b = blockIdx.x;
  const int tid = threadIdx.x;
  const int w = tid >> 5, lane = tid & 31;
  v8f z = {0, 0, 0, 0, 0, 0, 0, 0};

  // --- phase 0a: zero pool, de[e] = He[b,e,:].p_v ------------------------
  if (tid < H_) pool_s[tid] = 0.f;
  for (int e = tid; e < EPAD; e += 256) {
    float s = 0.f;
    if (e < E_) {
      const float* r = Hein + ((size_t)b * E_ + e) * IN_E;
#pragma unroll
      for (int f = 0; f < IN_E; ++f) s += r[f] * pv[f];
    }
    de_s[e] = s;
  }

  // --- phase 0b: stage Hv[b] and W_v as f16 ------------------------------
  _Float16* Hv16 = (_Float16*)smem;            // [128][128], 32 KB
  _Float16* Wv16 = (_Float16*)(smem + 32768);  // [128][64],  16 KB
  for (int idx = tid; idx < 128 * 128; idx += 256) {
    int r = idx >> 7, k = idx & 127;
    Hv16[idx] = (_Float16)(
        (r < N_ && k < IN_V) ? Hvin[((size_t)b * N_ + r) * IN_V + k] : 0.f);
  }
  for (int idx = tid; idx < 128 * 64; idx += 256) {
    int k = idx >> 6, h = idx & 63;
    Wv16[idx] = (_Float16)((k < IN_V) ? Wv[(size_t)k * H_ + h] : 0.f);
  }
  __syncthreads();

  // --- phase 0c: HvW = Hv @ W_v (wave w owns 16-row tile w) --------------
  v8f cw[4];
#pragma unroll
  for (int ct = 0; ct < 4; ++ct) cw[ct] = z;
#pragma unroll
  for (int k0 = 0; k0 < 128; k0 += 32) {
    v16h a = load_a_frag(Hv16, w * 16, k0, 128);
#pragma unroll
    for (int ct = 0; ct < 4; ++ct) {
      v16h bf = load_b_frag_KN(Wv16, ct * 16, k0, 64);
      cw[ct] = __builtin_amdgcn_wmma_f32_16x16x32_f16(false, a, false, bf,
                                                      (short)0, cw[ct], false,
                                                      false);
    }
  }
  __syncthreads();  // all reads of Hv16/Wv16 done; both regions reusable

  // --- phase 0d: spill HvW to LDS f16 [128][64] at +32 KB ----------------
  _Float16* HvWs = (_Float16*)(smem + 32768);  // persists through phase 2
#pragma unroll
  for (int ct = 0; ct < 4; ++ct)
#pragma unroll
    for (int j = 0; j < 8; ++j)
      HvWs[(size_t)(w * 16 + j + ((lane >> 4) << 3)) * 64 + ct * 16 +
           (lane & 15)] = (_Float16)cw[ct][j];

  // --- phase 1: mult1 = (T*diag(de)) @ T^T, double-buffered chunks -------
  // Ping-pong buffers occupy [0,32K) (old Hv16 region). Each iteration:
  // store regs->buf[ch&1], issue global loads for ch+1 (latency hidden
  // behind barrier+WMMA), one barrier, 8 WMMAs. buf[ch&1] is re-written at
  // ch+2, always separated from its readers by the barrier at ch+1.
  const float* Tbat = T + (size_t)b * N_ * E_;
  float tv[16];
  {
#pragma unroll
    for (int i = 0; i < 16; ++i) {
      int idx = tid + i * 256;
      int m = idx >> 5, ge = idx & 31;  // chunk 0
      tv[i] = (m < N_ && ge < E_) ? Tbat[(size_t)m * E_ + ge] : 0.f;
    }
  }
  v8f acc[8];
#pragma unroll
  for (int t = 0; t < 8; ++t) acc[t] = z;

  for (int ch = 0; ch < NCH; ++ch) {
    _Float16* Ta = (_Float16*)(smem + (size_t)(ch & 1) * 16384);  // [128][32]
    _Float16* Tb = Ta + 128 * 32;                                 // [128][32]
#pragma unroll
    for (int i = 0; i < 16; ++i) {
      int idx = tid + i * 256;
      int ge = ch * 32 + (idx & 31);
      Ta[idx] = (_Float16)(tv[i] * de_s[ge]);  // A = T * diag(de)
      Tb[idx] = (_Float16)tv[i];               // B = T (read transposed)
    }
    if (ch + 1 < NCH) {
#pragma unroll
      for (int i = 0; i < 16; ++i) {
        int idx = tid + i * 256;
        int m = idx >> 5, ge = (ch + 1) * 32 + (idx & 31);
        tv[i] = (m < N_ && ge < E_) ? Tbat[(size_t)m * E_ + ge] : 0.f;
      }
    }
    __syncthreads();
    v16h a = load_a_frag(Ta, w * 16, 0, 32);
#pragma unroll
    for (int t = 0; t < 8; ++t) {
      v16h bf = load_b_frag_NK(Tb, t * 16, 32);
      acc[t] = __builtin_amdgcn_wmma_f32_16x16x32_f16(false, a, false, bf,
                                                      (short)0, acc[t], false,
                                                      false);
    }
  }
  __syncthreads();  // last chunk consumed; [0,32K) reusable for adjA

  // --- adjA = (eye ? 1 : mult1) * Av, f16 [128][128], zero-padded --------
  _Float16* adjA = (_Float16*)smem;
#pragma unroll
  for (int t = 0; t < 8; ++t) {
#pragma unroll
    for (int j = 0; j < 8; ++j) {
      int n = w * 16 + j + ((lane >> 4) << 3);
      int m = t * 16 + (lane & 15);
      float v = 0.f;
      if (n < N_ && m < N_) {
        float m1 = (n == m) ? 1.f : acc[t][j];
        v = m1 * Av[((size_t)b * N_ + n) * N_ + m];
      }
      adjA[(size_t)n * 128 + m] = (_Float16)v;
    }
  }
  __syncthreads();

  // --- phase 2: Hv2 = leaky(adjA @ HvW + b_v); pool over nodes -----------
#pragma unroll
  for (int ct = 0; ct < 4; ++ct) {
    v8f c = z;
#pragma unroll
    for (int k0 = 0; k0 < 128; k0 += 32) {
      v16h a = load_a_frag(adjA, w * 16, k0, 128);
      v16h bf = load_b_frag_KN(HvWs, ct * 16, k0, 64);
      c = __builtin_amdgcn_wmma_f32_16x16x32_f16(false, a, false, bf, (short)0,
                                                 c, false, false);
    }
    int h = ct * 16 + (lane & 15);
    float bias = bv[h];
    float s = 0.f;
#pragma unroll
    for (int j = 0; j < 8; ++j) {
      int n = w * 16 + j + ((lane >> 4) << 3);
      float v = c[j] + bias;
      v = (v > 0.f) ? v : NEG_SLOPE * v;
      if (n < N_) s += v;  // rows >= 116 are padding
    }
    s += __shfl_xor(s, 16, 32);  // merge the two M-halves (same h)
    if (lane < 16) atomicAdd(&pool_s[h], s);  // ds_add_f32, intra-WGP
  }
  __syncthreads();

  // --- classifier head: out[b,:] = (pool/N) @ W_fc + b_fc ----------------
  if (tid < OUT_) {
    float s = 0.f;
#pragma unroll
    for (int h = 0; h < H_; ++h) s += pool_s[h] * Wfc[(size_t)h * OUT_ + tid];
    out[(size_t)b * OUT_ + tid] = s * (1.f / (float)N_) + bfc[tid];
  }
}

extern "C" void kernel_launch(void* const* d_in, const int* in_sizes, int n_in,
                              void* d_out, int out_size, void* d_ws,
                              size_t ws_size, hipStream_t stream) {
  (void)in_sizes; (void)n_in; (void)out_size; (void)d_ws; (void)ws_size;
  const float* H_v   = (const float*)d_in[0];
  const float* H_e   = (const float*)d_in[1];
  const float* adj_v = (const float*)d_in[2];
  // d_in[3] (adj_e) and d_in[8..10] (W_e, p_e, b_e) feed only the edge
  // layer, which never reaches the returned tensor -> never read
  // (saves ~232 MB of HBM traffic and ~21 of the 22 GFLOP).
  const float* T     = (const float*)d_in[4];
  const float* W_v   = (const float*)d_in[5];
  const float* p_v   = (const float*)d_in[6];
  const float* b_v   = (const float*)d_in[7];
  const float* W_fc  = (const float*)d_in[11];
  const float* b_fc  = (const float*)d_in[12];

  k_fused<<<B_, 256, 0, stream>>>(H_v, H_e, adj_v, T, W_v, p_v, b_v, W_fc,
                                  b_fc, (float*)d_out);
}
